// Attention_79362405696058
// MI455X (gfx1250) — compile-verified, matrix operand order checked by reading, not run on hardware
//
#include <hip/hip_runtime.h>

typedef __attribute__((ext_vector_type(4)))  _Float16 h4;
typedef __attribute__((ext_vector_type(8)))  _Float16 h8;
typedef __attribute__((ext_vector_type(16))) _Float16 h16;
typedef __attribute__((ext_vector_type(8)))  float    f8;

#define S_LEN 2048
#define HID   2048
#define NH    32
#define NKV   8
#define HD    64

// ---------------------------------------------------------------------------
// CDNA5 async global->LDS copy (16B per lane), tracked by ASYNCcnt.
// LDS operand is the low 32 bits of the generic pointer (LDS aperture
// truncates to addr[31:0], ISA 10.2).
// ---------------------------------------------------------------------------
__device__ __forceinline__ void async_copy_b128(void* lds_dst, const void* gsrc) {
  unsigned lds_off = (unsigned)(unsigned long long)lds_dst;
  unsigned long long ga = (unsigned long long)gsrc;
  asm volatile("global_load_async_to_lds_b128 %0, %1, off"
               :: "v"(lds_off), "v"(ga) : "memory");
}
__device__ __forceinline__ void wait_async0() {
#if defined(__has_builtin) && __has_builtin(__builtin_amdgcn_s_wait_asynccnt)
  __builtin_amdgcn_s_wait_asynccnt(0);
#else
  asm volatile("s_wait_asynccnt 0" ::: "memory");
#endif
}

// ---------------------------------------------------------------------------
// f32 -> f16 convert (vectorized x4)
// ---------------------------------------------------------------------------
__global__ void cvt_f32_f16(const float* __restrict__ in, _Float16* __restrict__ out, int n4) {
  int idx = blockIdx.x * blockDim.x + threadIdx.x;
  if (idx >= n4) return;
  float4 v = ((const float4*)in)[idx];
  h4 o;
  o[0] = (_Float16)v.x; o[1] = (_Float16)v.y;
  o[2] = (_Float16)v.z; o[3] = (_Float16)v.w;
  ((h4*)out)[idx] = o;
}

// ---------------------------------------------------------------------------
// RoPE (f32 in) -> f16 out, scale folds 1/sqrt(HD) into Q.
// layout: [S][heads][64]; thread handles pair (i, i+32), i in 0..31
// ---------------------------------------------------------------------------
__global__ void rope_f16(const float* __restrict__ in, _Float16* __restrict__ out,
                         int heads, float scale, int total) {
  int idx = blockIdx.x * blockDim.x + threadIdx.x;
  if (idx >= total) return;
  int i   = idx & 31;
  int rem = idx >> 5;
  int hh  = rem % heads;
  int s   = rem / heads;
  size_t base = ((size_t)s * heads + hh) * HD;
  float x1 = in[base + i];
  float x2 = in[base + i + 32];
  float ang = (float)s * __powf(10000.0f, -(float)i * (1.0f / 32.0f));
  float sn, cs;
  __sincosf(ang, &sn, &cs);
  out[base + i]      = (_Float16)((x1 * cs - x2 * sn) * scale);
  out[base + i + 32] = (_Float16)((x2 * cs + x1 * sn) * scale);
}

// ---------------------------------------------------------------------------
// GEMM: C[M x N] (f32) = A[M x K] (f16) * B[K x N] (f16)
// block = 256 thr (8 waves as 4M x 2N); block tile 128M x 128N;
// wave tile 32M x 64N -> 8 WMMA per 32-K step.
// B tile (32K x 128N) staged transposed into padded LDS; each thread loads
// two adjacent K rows and writes packed dwords (8 x ds_store_b32).
// Fragment reads are ds_load_b128; A frags are contiguous 16B global loads.
// ---------------------------------------------------------------------------
__global__ __launch_bounds__(256) void gemm_f16f32(
    const _Float16* __restrict__ A, int lda,
    const _Float16* __restrict__ B, int ldb,
    float* __restrict__ C, int ldc, int K) {
  constexpr int BT_LD = 40;  // 32 + 8 pad (halves)
  __shared__ __align__(16) _Float16 BT[128 * BT_LD];  // 10 KB

  const int tid  = threadIdx.x;
  const int lane = tid & 31;
  const int wave = tid >> 5;
  const int hh   = lane >> 4;
  const int ln   = lane & 15;
  const int wm   = wave & 3;    // 4 waves along M
  const int wn   = wave >> 2;   // 2 waves along N
  const int m0   = blockIdx.y * 128 + wm * 32;
  const int n0   = blockIdx.x * 128;
  const int nw0  = wn * 64;

  const int srow = (tid >> 4) * 2;  // 0,2,..,30 (even K row)
  const int scol = (tid & 15) * 8;  // 0..120

  f8 acc[2][4] = {};

  for (int k0 = 0; k0 < K; k0 += 32) {
    __syncthreads();
    // stage B[32 x 128] transposed into LDS; pack adjacent K-rows as dwords
    {
      const _Float16* bp0 = B + (size_t)(k0 + srow) * ldb + n0 + scol;
      h8 b0 = *(const h8*)bp0;
      h8 b1 = *(const h8*)(bp0 + ldb);
#pragma unroll
      for (int i = 0; i < 8; ++i) {
        union { _Float16 hv[2]; unsigned u; } p;
        p.hv[0] = b0[i]; p.hv[1] = b1[i];
        *(unsigned*)(BT + (scol + i) * BT_LD + srow) = p.u;
      }
    }
    if (k0 + 32 < K)
      __builtin_prefetch(B + (size_t)(k0 + 32 + srow) * ldb + n0 + scol, 0, 1);
    __syncthreads();

    // A fragments: two 16-row chunks, K chunk k0..k0+31
    h16 af[2];
#pragma unroll
    for (int mi = 0; mi < 2; ++mi) {
      const _Float16* ap = A + (size_t)(m0 + mi * 16 + ln) * lda + k0;
      h8 alo = *(const h8*)(ap + hh * 8);
      h8 ahi = *(const h8*)(ap + 16 + hh * 8);
#pragma unroll
      for (int i = 0; i < 8; ++i) { af[mi][i] = alo[i]; af[mi][8 + i] = ahi[i]; }
    }

#pragma unroll
    for (int nc = 0; nc < 4; ++nc) {
      const _Float16* bp = BT + (nw0 + nc * 16 + ln) * BT_LD;
      h8 blo = *(const h8*)(bp + hh * 8);
      h8 bhi = *(const h8*)(bp + 16 + hh * 8);
      h16 bf;
#pragma unroll
      for (int i = 0; i < 8; ++i) { bf[i] = blo[i]; bf[8 + i] = bhi[i]; }
#pragma unroll
      for (int mi = 0; mi < 2; ++mi)
        acc[mi][nc] = __builtin_amdgcn_wmma_f32_16x16x32_f16(
            false, af[mi], false, bf, (short)0, acc[mi][nc], false, false);
    }
  }

#pragma unroll
  for (int mi = 0; mi < 2; ++mi)
#pragma unroll
    for (int nc = 0; nc < 4; ++nc)
#pragma unroll
      for (int j = 0; j < 8; ++j)
        C[(size_t)(m0 + mi * 16 + hh * 8 + j) * ldc + (n0 + nw0 + nc * 16 + ln)] =
            acc[mi][nc][j];
}

// ---------------------------------------------------------------------------
// Flash attention (causal, GQA). block = 256 thr (8 waves), q rows
// [Q0, Q0+128) of one head; each wave owns 16 q rows. K/V tiles shared by
// all waves: K staged via async global->LDS DMA (row-major), V staged
// transposed with packed dword stores. Scores computed transposed
// (S^T = K * Q^T): the f32 C-frag layout of S^T equals the f16 B-frag layout
// of P^T for out^T = V^T * P^T, so softmax stays in-register (xor-16
// shuffles only). All waves iterate the block-max kv count (uniform
// barriers); waves past their causal bound skip compute.
// ---------------------------------------------------------------------------
__global__ __launch_bounds__(256) void attn_flash(
    const _Float16* __restrict__ Qh,   // [S][NH*HD], pre-scaled by 1/sqrt(HD)
    const _Float16* __restrict__ Kh,   // [S][NKV*HD]
    const _Float16* __restrict__ Vh,   // [S][NKV*HD]
    _Float16* __restrict__ Oh) {       // [S][NH*HD]
  constexpr int VT_LD = 40;  // 32 + 8 pad
  __shared__ __align__(16) _Float16 Kt[32 * HD];       // 4 KB, row-major
  __shared__ __align__(16) _Float16 Vt[HD * VT_LD];    // 5 KB, transposed

  const int tid  = threadIdx.x;
  const int lane = tid & 31;
  const int wave = tid >> 5;
  const int hh   = lane >> 4;
  const int ln   = lane & 15;
  const int head = blockIdx.y;
  const int kvh  = head & (NKV - 1);
  const int Q0   = blockIdx.x * 128;
  const int q0   = Q0 + wave * 16;
  const int qi   = q0 + ln;            // this lane's query column

  const int krow = tid >> 3;           // 0..31 (K staging row)
  const int kcol = (tid & 7) * 8;      // 0..56

  // Hoisted Q^T B-fragments (2 chunks over HD=64)
  const _Float16* qp = Qh + (size_t)qi * (NH * HD) + head * HD;
  h16 qb[2];
#pragma unroll
  for (int c = 0; c < 2; ++c) {
    h8 lo = *(const h8*)(qp + c * 32 + hh * 8);
    h8 hi = *(const h8*)(qp + c * 32 + 16 + hh * 8);
#pragma unroll
    for (int i = 0; i < 8; ++i) { qb[c][i] = lo[i]; qb[c][8 + i] = hi[i]; }
  }

  f8 acc[4] = {};                      // out^T accum: 4 d-chunks of 16
  float m_run = -1e30f, s_run = 0.0f;

  const int nblk_max = (Q0 >> 5) + 4;      // block-uniform kv tile count
  const int nblk_w   = (q0 + 47) >> 5;     // this wave's causal bound

  for (int kb = 0; kb < nblk_max; ++kb) {
    const int k0 = kb * 32;
    __syncthreads();

    // K tile: async DMA global->LDS, 16B per lane (tracked by ASYNCcnt)
    async_copy_b128(Kt + krow * HD + kcol,
                    Kh + (size_t)(k0 + krow) * (NKV * HD) + kvh * HD + kcol);

    // V tile transposed: threads 0..127, two adjacent kv rows packed to dwords
    if (tid < 128) {
      const int vr = (tid >> 3) * 2;       // 0,2,..,30
      const int vc = (tid & 7) * 8;        // 0..56
      const _Float16* vp0 =
          Vh + (size_t)(k0 + vr) * (NKV * HD) + kvh * HD + vc;
      h8 v0 = *(const h8*)vp0;
      h8 v1 = *(const h8*)(vp0 + NKV * HD);
#pragma unroll
      for (int i = 0; i < 8; ++i) {
        union { _Float16 hv[2]; unsigned u; } p;
        p.hv[0] = v0[i]; p.hv[1] = v1[i];
        *(unsigned*)(Vt + (vc + i) * VT_LD + vr) = p.u;
      }
    }
    wait_async0();
    __syncthreads();

    if (kb >= nblk_w) continue;  // past causal range for this wave

    // S^T = K_tile(32x64) * Q^T(64x16): two 16x16 f32 frags
    f8 sc[2];
#pragma unroll
    for (int g = 0; g < 2; ++g) {
      f8 z = {};
#pragma unroll
      for (int c = 0; c < 2; ++c) {
        const _Float16* kp = Kt + (g * 16 + ln) * HD + c * 32;
        h8 lo = *(const h8*)(kp + hh * 8);
        h8 hi = *(const h8*)(kp + 16 + hh * 8);
        h16 ka;
#pragma unroll
        for (int i = 0; i < 8; ++i) { ka[i] = lo[i]; ka[8 + i] = hi[i]; }
        z = __builtin_amdgcn_wmma_f32_16x16x32_f16(
            false, ka, false, qb[c], (short)0, z, false, false);
      }
      sc[g] = z;
    }

    // causal mask + online softmax (fixed q col per lane)
    float lmax = -1e30f;
#pragma unroll
    for (int g = 0; g < 2; ++g)
#pragma unroll
      for (int j = 0; j < 8; ++j) {
        int ki  = k0 + g * 16 + hh * 8 + j;
        float s = (ki <= qi) ? sc[g][j] : -1e9f;
        sc[g][j] = s;
        lmax = fmaxf(lmax, s);
      }
    lmax = fmaxf(lmax, __shfl_xor(lmax, 16, 32));
    float m_new   = fmaxf(m_run, lmax);
    float rescale = __expf(m_run - m_new);
    float lsum = 0.0f;
    h16 pb;                            // P^T as B-fragment (K=32)
#pragma unroll
    for (int g = 0; g < 2; ++g)
#pragma unroll
      for (int j = 0; j < 8; ++j) {
        float p = __expf(sc[g][j] - m_new);
        lsum += p;
        pb[g * 8 + j] = (_Float16)p;
      }
    lsum += __shfl_xor(lsum, 16, 32);
    s_run = s_run * rescale + lsum;
    m_run = m_new;
#pragma unroll
    for (int dc = 0; dc < 4; ++dc)
#pragma unroll
      for (int j = 0; j < 8; ++j) acc[dc][j] *= rescale;

    // out^T += V^T(64x32) * P^T(32x16), V^T frags from transposed LDS
#pragma unroll
    for (int dc = 0; dc < 4; ++dc) {
      const _Float16* vp = Vt + (dc * 16 + ln) * VT_LD;
      h8 lo = *(const h8*)(vp + hh * 8);
      h8 hi = *(const h8*)(vp + 16 + hh * 8);
      h16 va;
#pragma unroll
      for (int i = 0; i < 8; ++i) { va[i] = lo[i]; va[8 + i] = hi[i]; }
      acc[dc] = __builtin_amdgcn_wmma_f32_16x16x32_f16(
          false, va, false, pb, (short)0, acc[dc], false, false);
    }
  }

  // normalize + store f16 (feeds the Wo GEMM); 16B stores
  float inv = 1.0f / s_run;
  _Float16* op = Oh + (size_t)qi * (NH * HD) + head * HD;
#pragma unroll
  for (int dc = 0; dc < 4; ++dc) {
    h8 o;
#pragma unroll
    for (int j = 0; j < 8; ++j) o[j] = (_Float16)(acc[dc][j] * inv);
    *(h8*)(op + dc * 16 + hh * 8) = o;
  }
}

// ---------------------------------------------------------------------------
// host orchestration
// ---------------------------------------------------------------------------
extern "C" void kernel_launch(void* const* d_in, const int* in_sizes, int n_in,
                              void* d_out, int out_size, void* d_ws, size_t ws_size,
                              hipStream_t stream) {
  (void)in_sizes; (void)n_in; (void)out_size; (void)ws_size;
  const float* hs = (const float*)d_in[0];
  // d_in[1] = attention_mask: causal, computed analytically in-kernel
  const float* Wq = (const float*)d_in[2];
  const float* Wk = (const float*)d_in[3];
  const float* Wv = (const float*)d_in[4];
  const float* Wo = (const float*)d_in[5];

  char* ws = (char*)d_ws;
  size_t off = 0;
  auto alloc = [&](size_t bytes) {
    char* p = ws + off;
    off += (bytes + 255) & ~(size_t)255;
    return p;
  };
  _Float16* hs_h = (_Float16*)alloc((size_t)S_LEN * HID * 2);
  _Float16* wq_h = (_Float16*)alloc((size_t)HID * NH * HD * 2);
  _Float16* wk_h = (_Float16*)alloc((size_t)HID * NKV * HD * 2);
  _Float16* wv_h = (_Float16*)alloc((size_t)HID * NKV * HD * 2);
  _Float16* wo_h = (_Float16*)alloc((size_t)HID * HID * 2);
  float*    q_f  = (float*)   alloc((size_t)S_LEN * NH * HD * 4);
  float*    k_f  = (float*)   alloc((size_t)S_LEN * NKV * HD * 4);
  float*    v_f  = (float*)   alloc((size_t)S_LEN * NKV * HD * 4);
  _Float16* q_h  = (_Float16*)alloc((size_t)S_LEN * NH * HD * 2);
  _Float16* k_h  = (_Float16*)alloc((size_t)S_LEN * NKV * HD * 2);
  _Float16* v_h  = (_Float16*)alloc((size_t)S_LEN * NKV * HD * 2);
  _Float16* ao_h = (_Float16*)alloc((size_t)S_LEN * NH * HD * 2);

  const int TB = 256;
  auto blks = [](int n, int b) { return (n + b - 1) / b; };

  // 1) down-convert operands to f16
  cvt_f32_f16<<<blks(S_LEN * HID / 4, TB), TB, 0, stream>>>(hs, hs_h, S_LEN * HID / 4);
  cvt_f32_f16<<<blks(HID * NH * HD / 4, TB), TB, 0, stream>>>(Wq, wq_h, HID * NH * HD / 4);
  cvt_f32_f16<<<blks(HID * NKV * HD / 4, TB), TB, 0, stream>>>(Wk, wk_h, HID * NKV * HD / 4);
  cvt_f32_f16<<<blks(HID * NKV * HD / 4, TB), TB, 0, stream>>>(Wv, wv_h, HID * NKV * HD / 4);
  cvt_f32_f16<<<blks(HID * HID / 4, TB), TB, 0, stream>>>(Wo, wo_h, HID * HID / 4);

  // 2) QKV projections (WMMA), block tile 128x128
  dim3 gq(NH * HD / 128, S_LEN / 128);
  dim3 gkv(NKV * HD / 128, S_LEN / 128);
  gemm_f16f32<<<gq, TB, 0, stream>>>(hs_h, HID, wq_h, NH * HD, q_f, NH * HD, HID);
  gemm_f16f32<<<gkv, TB, 0, stream>>>(hs_h, HID, wk_h, NKV * HD, k_f, NKV * HD, HID);
  gemm_f16f32<<<gkv, TB, 0, stream>>>(hs_h, HID, wv_h, NKV * HD, v_f, NKV * HD, HID);

  // 3) RoPE (+fold softmax scale into Q); plain convert for V
  int qtot = S_LEN * NH * 32, ktot = S_LEN * NKV * 32;
  rope_f16<<<blks(qtot, TB), TB, 0, stream>>>(q_f, q_h, NH, 0.125f, qtot);
  rope_f16<<<blks(ktot, TB), TB, 0, stream>>>(k_f, k_h, NKV, 1.0f, ktot);
  cvt_f32_f16<<<blks(S_LEN * NKV * HD / 4, TB), TB, 0, stream>>>(v_f, v_h, S_LEN * NKV * HD / 4);

  // 4) causal flash attention
  dim3 ga(S_LEN / 128, NH);
  attn_flash<<<ga, TB, 0, stream>>>(q_h, k_h, v_h, ao_h);

  // 5) output projection -> fp32 d_out
  dim3 go(HID / 128, S_LEN / 128);
  gemm_f16f32<<<go, TB, 0, stream>>>(ao_h, NH * HD, wo_h, HID, (float*)d_out, HID, HID);
}
